// DeformableHead_46961172415018
// MI455X (gfx1250) — compile-verified, matrix operand order checked by reading, not run on hardware
//
#include <hip/hip_runtime.h>
#include <hip/hip_bf16.h>
#include <math.h>

// ---------------------------------------------------------------------------
// Types for CDNA5 WMMA (wave32, 16x16x32 bf16 -> f32)
// ---------------------------------------------------------------------------
typedef __attribute__((ext_vector_type(16))) __bf16 v16bf;
typedef __attribute__((ext_vector_type(8)))  float  v8f;

#define DD 256
#define NQ 200
#define NP 8
#define PS 3
#define NL 6
#define BB 4

// Load 8+8 contiguous floats (k = base+0..7 and base+16..23) as 4x b128.
__device__ __forceinline__ void load8x2(const float* __restrict__ p, float* d) {
  float4 v0 = *(const float4*)(p);
  float4 v1 = *(const float4*)(p + 4);
  float4 v2 = *(const float4*)(p + 16);
  float4 v3 = *(const float4*)(p + 20);
  d[0] = v0.x; d[1] = v0.y; d[2]  = v0.z; d[3]  = v0.w;
  d[4] = v1.x; d[5] = v1.y; d[6]  = v1.z; d[7]  = v1.w;
  d[8] = v2.x; d[9] = v2.y; d[10] = v2.z; d[11] = v2.w;
  d[12] = v3.x; d[13] = v3.y; d[14] = v3.z; d[15] = v3.w;
}

__device__ __forceinline__ v16bf pack_bf16(const float* f) {
  v16bf r;
#pragma unroll
  for (int h = 0; h < 16; ++h) r[h] = (__bf16)f[h];
  return r;
}

__device__ __forceinline__ float epilogue_act(float v, int act) {
  if (act == 1)      v = fmaxf(v, 0.f);
  else if (act == 2) v = tanhf(v);
  else if (act == 3) v = 1.f / (1.f + __expf(-v));
  return v;
}

// ---------------------------------------------------------------------------
// BN fold: scale = g*rsqrt(v+eps); bias = (conv_b - m)*scale + beta
// ---------------------------------------------------------------------------
__global__ void bnprep_k(const float* __restrict__ g, const float* __restrict__ be,
                         const float* __restrict__ mn, const float* __restrict__ vr,
                         const float* __restrict__ cb,
                         float* __restrict__ sc, float* __restrict__ bi, int C) {
  int i = blockIdx.x * blockDim.x + threadIdx.x;
  if (i >= C) return;
  float s = g[i] * rsqrtf(vr[i] + 1e-5f);
  sc[i] = s;
  bi[i] = (cb[i] - mn[i]) * s + be[i];
}

// ---------------------------------------------------------------------------
// conv1: 7x7 stride2 pad3, Cin=3 -> 64, fused BN+ReLU (Cin too small for WMMA)
// ---------------------------------------------------------------------------
__global__ void conv1_k(const float* __restrict__ x, const float* __restrict__ w,
                        const float* __restrict__ sc, const float* __restrict__ bi,
                        float* __restrict__ out) {
  int idx = blockIdx.x * blockDim.x + threadIdx.x;
  if (idx >= BB * 64 * 256 * 256) return;
  int ow = idx & 255; int t = idx >> 8;
  int oh = t & 255;   t >>= 8;
  int co = t & 63;    int b = t >> 6;
  const float* wc = w + co * 3 * 49;
  float acc = 0.f;
  for (int ci = 0; ci < 3; ++ci)
    for (int kh = 0; kh < 7; ++kh) {
      int ih = oh * 2 - 3 + kh;
      if ((unsigned)ih >= 512u) continue;
      const float* xr = x + (((size_t)b * 3 + ci) * 512 + ih) * 512;
      const float* wr = wc + ci * 49 + kh * 7;
      for (int kw = 0; kw < 7; ++kw) {
        int iw = ow * 2 - 3 + kw;
        if ((unsigned)iw >= 512u) continue;
        acc = fmaf(xr[iw], wr[kw], acc);
      }
    }
  float v = acc * sc[co] + bi[co];
  out[idx] = fmaxf(v, 0.f);
}

// ---------------------------------------------------------------------------
// Implicit-GEMM conv via WMMA bf16 (KS compile-time: 1 or 3).
// M = Cout (tile 16), N = 16 output pixels, K = Cin*KS*KS (multiple of 32).
// Weight rows are contiguous -> b128 loads. Input is a bounds-checked gather.
// ---------------------------------------------------------------------------
template <int KS>
__global__ void conv_wmma_k(const float* __restrict__ in, const float* __restrict__ w,
                            const float* __restrict__ sc, const float* __restrict__ bi,
                            float* __restrict__ out,
                            int Cin, int Hin, int Win, int Cout, int Hout, int Wout,
                            int stride, int pad, int relu) {
  constexpr int KSQ = KS * KS;
  int wave = (blockIdx.x * blockDim.x + threadIdx.x) >> 5;
  int lane = threadIdx.x & 31;
  int tilesM = Cout >> 4;
  int tilesN = (BB * Hout * Wout) >> 4;
  if (wave >= tilesM * tilesN) return;
  int tm = wave % tilesM, tn = wave / tilesM;
  int K = Cin * KSQ;

  int m   = (tm << 4) + (lane & 15);            // output channel (A row)
  int pix = (tn << 4) + (lane & 15);            // output pixel  (B col)
  int b   = pix / (Hout * Wout);
  int r2  = pix % (Hout * Wout);
  int oh  = r2 / Wout, ow = r2 % Wout;
  int kOff = (lane >> 4) << 3;
  const float* arow = w + (size_t)m * K;
  int ihBase = oh * stride - pad;
  int iwBase = ow * stride - pad;

  v8f acc = {};
  for (int k0 = 0; k0 < K; k0 += 32) {
    int kb = k0 + kOff;
    float af[16], bfv[16];
    load8x2(arow + kb, af);                     // weights: contiguous, in-bounds
#pragma unroll
    for (int h = 0; h < 16; ++h) {
      int k = kb + (h < 8 ? h : h + 8);
      int ci = k / KSQ; int rr = k - ci * KSQ;  // KSQ is constexpr -> cheap
      int kh = rr / KS, kw = rr - kh * KS;
      int ih = ihBase + kh, iw = iwBase + kw;
      float v = 0.f;
      if ((unsigned)ih < (unsigned)Hin && (unsigned)iw < (unsigned)Win)
        v = in[(((size_t)b * Cin + ci) * Hin + ih) * Win + iw];
      bfv[h] = v;
    }
    v16bf a = pack_bf16(af), bf = pack_bf16(bfv);
    acc = __builtin_amdgcn_wmma_f32_16x16x32_bf16(false, a, false, bf, (short)0, acc,
                                                  false, false);
  }
#pragma unroll
  for (int r = 0; r < 8; ++r) {
    int co = (tm << 4) + r + ((lane >> 4) << 3);
    float v = acc[r];
    v = v * (sc ? sc[co] : 1.f) + (bi ? bi[co] : 0.f);
    if (relu) v = fmaxf(v, 0.f);
    out[(((size_t)b * Cout + co) * Hout + oh) * Wout + ow] = v;
  }
}

// ---------------------------------------------------------------------------
// Sine positional encoding (d=256, Hf=Wf=64), added to h3 for all batches.
// ---------------------------------------------------------------------------
__global__ void posenc_k(float* __restrict__ h) {
  int idx = blockIdx.x * blockDim.x + threadIdx.x;
  if (idx >= 256 * 64 * 64) return;
  int xC = idx & 63; int t = idx >> 6;
  int yC = t & 63;   int c = t >> 6;
  bool isY = c < 128;
  int cc = isY ? c : c - 128;
  int j = cc >> 1;
  float dim_t = __powf(10000.f, (float)(2 * (j >> 1)) * (1.f / 64.f));
  float e = (isY ? (float)(yC + 1) : (float)(xC + 1)) * (6.28318530718f / (64.f + 1e-6f));
  float arg = e / dim_t;
  float v = (cc & 1) ? __cosf(arg) : __sinf(arg);
  for (int b = 0; b < BB; ++b)
    h[(((size_t)b * 256 + c) * 64 + yC) * 64 + xC] += v;
}

// ---------------------------------------------------------------------------
// Batched GEMM via WMMA bf16:  C = act(alpha * A @ op(B) + bias)
//   A: [M,K] row-major fp32;  B: transB ? [N,K] : [K,N]
// ALIGNED=1: M%16==0, N%16==0, K%32==0, transB==1 -> branch-free inner loop,
//            pure b128 loads, unconditional epilogue (all heavy decoder GEMMs).
// ALIGNED=0: fully bounds-checked generic path (ragged shapes only).
// ---------------------------------------------------------------------------
template <int ALIGNED>
__global__ void gemm_k(const float* __restrict__ A, const float* __restrict__ Bm,
                       const float* __restrict__ bias, float* __restrict__ C,
                       int M, int N, int K, int transB, int act, float alpha,
                       int nbatch, long sA, long sB, long sC) {
  int wave = (blockIdx.x * blockDim.x + threadIdx.x) >> 5;
  int lane = threadIdx.x & 31;
  int tilesM = (M + 15) >> 4, tilesN = (N + 15) >> 4;
  int perB = tilesM * tilesN;
  if (wave >= perB * nbatch) return;
  int bb = wave / perB; int tt = wave % perB;
  int tm = tt % tilesM, tn = tt / tilesM;
  const float* Ab = A  + (size_t)bb * sA;
  const float* Bb = Bm + (size_t)bb * sB;
  float*       Cb = C  + (size_t)bb * sC;

  int m = (tm << 4) + (lane & 15);
  int n = (tn << 4) + (lane & 15);
  int kOff = (lane >> 4) << 3;

  v8f acc = {};
  if (ALIGNED) {
    const float* pA = Ab + (size_t)m * K;       // in-bounds by contract
    const float* pB = Bb + (size_t)n * K;       // transB==1 by contract
    for (int k0 = 0; k0 < K; k0 += 32) {
      int kb = k0 + kOff;
      __builtin_prefetch(pA + kb + 256, 0, 1);  // global_prefetch_b8 (speculative)
      __builtin_prefetch(pB + kb + 256, 0, 1);
      float af[16], bfv[16];
      load8x2(pA + kb, af);
      load8x2(pB + kb, bfv);
      v16bf a = pack_bf16(af), bf = pack_bf16(bfv);
      acc = __builtin_amdgcn_wmma_f32_16x16x32_bf16(false, a, false, bf, (short)0, acc,
                                                    false, false);
    }
#pragma unroll
    for (int r = 0; r < 8; ++r) {
      int mo = (tm << 4) + r + ((lane >> 4) << 3);
      int no = (tn << 4) + (lane & 15);
      float v = acc[r] * alpha + (bias ? bias[no] : 0.f);
      Cb[(size_t)mo * N + no] = epilogue_act(v, act);
    }
  } else {
    for (int k0 = 0; k0 < K; k0 += 32) {
      int kb = k0 + kOff;
      float af[16], bfv[16];
      if (m < M && kb + 24 <= K) {
        load8x2(Ab + (size_t)m * K + kb, af);
      } else {
#pragma unroll
        for (int h = 0; h < 16; ++h) {
          int k = kb + (h < 8 ? h : h + 8);
          af[h] = (m < M && k < K) ? Ab[(size_t)m * K + k] : 0.f;
        }
      }
      if (transB && n < N && kb + 24 <= K) {
        load8x2(Bb + (size_t)n * K + kb, bfv);
      } else {
#pragma unroll
        for (int h = 0; h < 16; ++h) {
          int k = kb + (h < 8 ? h : h + 8);
          float bv = 0.f;
          if (n < N && k < K)
            bv = transB ? Bb[(size_t)n * K + k] : Bb[(size_t)k * N + n];
          bfv[h] = bv;
        }
      }
      v16bf a = pack_bf16(af), bf = pack_bf16(bfv);
      acc = __builtin_amdgcn_wmma_f32_16x16x32_bf16(false, a, false, bf, (short)0, acc,
                                                    false, false);
    }
#pragma unroll
    for (int r = 0; r < 8; ++r) {
      int mo = (tm << 4) + r + ((lane >> 4) << 3);
      int no = (tn << 4) + (lane & 15);
      if (mo < M && no < N) {
        float v = acc[r] * alpha + (bias ? bias[no] : 0.f);
        Cb[(size_t)mo * N + no] = epilogue_act(v, act);
      }
    }
  }
}

// ---------------------------------------------------------------------------
// Bilinear 3x3 patch sampling (grid_sample, border pad, align_corners=True).
// One block per (b,q,p,py,px); one thread per channel.
// s layout: [(b*NQ+q)*NP+p][ (py*PS+px)*DD + c ]   (K order matches pm1W)
// ---------------------------------------------------------------------------
__global__ void sample_k(const float* __restrict__ feat, const float* __restrict__ ro,
                         const float* __restrict__ boxes, float* __restrict__ s) {
  int blk = blockIdx.x;
  int pp = blk % (PS * PS); int t = blk / (PS * PS);
  int p = t % NP; t /= NP;
  int q = t % NQ; int b = t / NQ;
  int py = pp / PS, px = pp % PS;

  int bq = b * NQ + q;
  float rx = boxes[bq * 4 + 0] + 0.5f * ro[bq * (2 * NP) + 2 * p + 0];
  float ry = boxes[bq * 4 + 1] + 0.5f * ro[bq * (2 * NP) + 2 * p + 1];
  rx = fminf(fmaxf(rx, 0.f), 1.f);
  ry = fminf(fmaxf(ry, 0.f), 1.f);
  float ox = (float)(px - 1), oy = (float)(py - 1);
  float xs = fminf(fmaxf((rx + ox * (1.f / 64.f)) * 63.f, 0.f), 63.f);
  float ys = fminf(fmaxf((ry + oy * (1.f / 64.f)) * 63.f, 0.f), 63.f);
  int x0 = (int)floorf(xs), y0 = (int)floorf(ys);
  int x1 = min(x0 + 1, 63),  y1 = min(y0 + 1, 63);
  float wx = xs - (float)x0, wy = ys - (float)y0;

  int c = threadIdx.x;
  const float* fb = feat + ((size_t)b * DD + c) * 4096;
  float v00 = fb[y0 * 64 + x0], v01 = fb[y0 * 64 + x1];
  float v10 = fb[y1 * 64 + x0], v11 = fb[y1 * 64 + x1];
  float v = v00 * (1.f - wy) * (1.f - wx) + v01 * (1.f - wy) * wx +
            v10 * wy * (1.f - wx)         + v11 * wy * wx;
  s[((size_t)(bq * NP + p) * (PS * PS) + pp) * DD + c] = v;
}

// ---------------------------------------------------------------------------
// Small elementwise kernels
// ---------------------------------------------------------------------------
__global__ void qinit_k(const float* __restrict__ qe, const float* __restrict__ qp,
                        float* __restrict__ q) {
  int i = blockIdx.x * blockDim.x + threadIdx.x;
  if (i >= BB * NQ * DD) return;
  int r = i % (NQ * DD);
  q[i] = qe[r] + qp[r];
}

__global__ void add_k(float* __restrict__ a, const float* __restrict__ b, int n) {
  int i = blockIdx.x * blockDim.x + threadIdx.x;
  if (i < n) a[i] += b[i];
}

__global__ void copy_k(float* __restrict__ dst, const float* __restrict__ src, int n) {
  int i = blockIdx.x * blockDim.x + threadIdx.x;
  if (i < n) dst[i] = src[i];
}

__global__ void mean_k(const float* __restrict__ pv, float* __restrict__ kv) {
  int bq = blockIdx.x;              // 0..BB*NQ-1
  int c = threadIdx.x;              // 0..255
  float s = 0.f;
#pragma unroll
  for (int p = 0; p < NP; ++p) s += pv[((size_t)bq * NP + p) * DD + c];
  kv[(size_t)bq * DD + c] = s * (1.f / NP);
}

__global__ void refine_k(float* __restrict__ boxes, const float* __restrict__ delta, int n) {
  int i = blockIdx.x * blockDim.x + threadIdx.x;
  if (i >= n) return;
  float nb = boxes[i] + 0.1f * tanhf(delta[i] - 0.5f);
  boxes[i] = fminf(fmaxf(nb, 0.f), 1.f);
}

// ---------------------------------------------------------------------------
// Softmax over rows (one wave per row), wave32 shuffle reduction
// ---------------------------------------------------------------------------
__global__ void softmax_k(float* __restrict__ S, int rows, int cols) {
  int wave = (blockIdx.x * blockDim.x + threadIdx.x) >> 5;
  int lane = threadIdx.x & 31;
  if (wave >= rows) return;
  float* row = S + (size_t)wave * cols;
  float mx = -1e30f;
  for (int c = lane; c < cols; c += 32) mx = fmaxf(mx, row[c]);
  for (int o = 16; o > 0; o >>= 1) mx = fmaxf(mx, __shfl_xor(mx, o));
  float sum = 0.f;
  for (int c = lane; c < cols; c += 32) {
    float e = __expf(row[c] - mx);
    row[c] = e; sum += e;
  }
  for (int o = 16; o > 0; o >>= 1) sum += __shfl_xor(sum, o);
  float inv = 1.f / sum;
  for (int c = lane; c < cols; c += 32) row[c] *= inv;
}

// ---------------------------------------------------------------------------
// Host-side launch helpers
// ---------------------------------------------------------------------------
static inline void gemm(hipStream_t st, const float* A, const float* Bm,
                        const float* bias, float* C, int M, int N, int K,
                        int transB, int act, float alpha,
                        int nb = 1, long sA = 0, long sB = 0, long sC = 0) {
  int tiles = ((M + 15) / 16) * ((N + 15) / 16) * nb;
  int blocks = (tiles + 7) / 8;                 // 8 waves / block (wave32)
  bool aligned = (M % 16 == 0) && (N % 16 == 0) && (K % 32 == 0) && (transB == 1);
  if (aligned)
    gemm_k<1><<<blocks, 256, 0, st>>>(A, Bm, bias, C, M, N, K, transB, act, alpha,
                                      nb, sA, sB, sC);
  else
    gemm_k<0><<<blocks, 256, 0, st>>>(A, Bm, bias, C, M, N, K, transB, act, alpha,
                                      nb, sA, sB, sC);
}

static inline void conv_wmma(hipStream_t st, const float* in, const float* w,
                             const float* sc, const float* bi, float* out,
                             int Cin, int Hin, int Win, int Cout, int Hout, int Wout,
                             int ks, int stride, int pad, int relu) {
  int tiles = (Cout >> 4) * ((BB * Hout * Wout) >> 4);
  int blocks = (tiles + 7) / 8;
  if (ks == 1)
    conv_wmma_k<1><<<blocks, 256, 0, st>>>(in, w, sc, bi, out, Cin, Hin, Win,
                                           Cout, Hout, Wout, stride, pad, relu);
  else
    conv_wmma_k<3><<<blocks, 256, 0, st>>>(in, w, sc, bi, out, Cin, Hin, Win,
                                           Cout, Hout, Wout, stride, pad, relu);
}

extern "C" void kernel_launch(void* const* d_in, const int* in_sizes, int n_in,
                              void* d_out, int out_size, void* d_ws, size_t ws_size,
                              hipStream_t stream) {
  (void)in_sizes; (void)n_in; (void)out_size; (void)ws_size;
  const float* x    = (const float*)d_in[0];
  const float* pbi  = (const float*)d_in[1];
  const float* c1w = (const float*)d_in[2],  *c1b = (const float*)d_in[3];
  const float* bn1g = (const float*)d_in[4], *bn1b = (const float*)d_in[5];
  const float* bn1m = (const float*)d_in[6], *bn1v = (const float*)d_in[7];
  const float* c2w = (const float*)d_in[8],  *c2b = (const float*)d_in[9];
  const float* bn2g = (const float*)d_in[10], *bn2b = (const float*)d_in[11];
  const float* bn2m = (const float*)d_in[12], *bn2v = (const float*)d_in[13];
  const float* c3w = (const float*)d_in[14], *c3b = (const float*)d_in[15];
  const float* bn3g = (const float*)d_in[16], *bn3b = (const float*)d_in[17];
  const float* bn3m = (const float*)d_in[18], *bn3v = (const float*)d_in[19];
  const float* latw = (const float*)d_in[20], *latb = (const float*)d_in[21];
  const float* smw  = (const float*)d_in[22], *smb  = (const float*)d_in[23];
  const float* qemb = (const float*)d_in[24], *qpos = (const float*)d_in[25];
  const float* qW   = (const float*)d_in[26], *qb   = (const float*)d_in[27];
  const float* outW = (const float*)d_in[28], *outb = (const float*)d_in[29];
  const float* pm1W = (const float*)d_in[30], *pm1b = (const float*)d_in[31];
  const float* pm2W = (const float*)d_in[32], *pm2b = (const float*)d_in[33];
  const float* bh1W = (const float*)d_in[34], *bh1b = (const float*)d_in[35];
  const float* bh2W = (const float*)d_in[36], *bh2b = (const float*)d_in[37];
  const float* rp1W = (const float*)d_in[38], *rp1b = (const float*)d_in[39];
  const float* rp2W = (const float*)d_in[40], *rp2b = (const float*)d_in[41];
  const float* clsW = (const float*)d_in[42], *clsb = (const float*)d_in[43];
  float* out = (float*)d_out;

  // -------- workspace layout (floats) --------
  float* W = (float*)d_ws;
  float* sc1 = W;        float* bi1 = W + 64;
  float* sc2 = W + 128;  float* bi2 = W + 256;
  float* sc3 = W + 384;  float* bi3 = W + 640;
  float* h1   = W + 1024;            // 4*64*256*256  = 16,777,216
  float* h2   = h1 + 16777216;       // 4*128*128*128 =  8,388,608 (reused in decoder)
  float* h3   = h2 + 8388608;        // 4*256*64*64   =  4,194,304
  float* lat  = h3 + 4194304;
  float* feat = lat + 4194304;
  // decoder aliases (inside h1/h2 regions, backbone done by then)
  float* sbuf    = h1;               // 6400*2304 = 14,745,600
  float* t2      = h2;               // 6400*256
  float* pv      = t2 + 1638400;     // 6400*256
  float* queries = pv + 1638400;     // 800*256
  float* t1      = queries + 204800;
  float* kv      = t1 + 204800;
  float* qbuf    = kv + 204800;
  float* ctx     = qbuf + 204800;
  float* outbuf  = ctx + 204800;
  float* scores  = outbuf + 204800;  // 4*200*200
  float* ro      = scores + 160000;  // 800*16
  float* delta   = ro + 12800;       // 800*4
  float* boxes   = delta + 3200;     // 800*4

  const float scale = 0.0625f;       // 1/sqrt(256)

  // -------- backbone --------
  bnprep_k<<<1, 64, 0, stream>>>(bn1g, bn1b, bn1m, bn1v, c1b, sc1, bi1, 64);
  bnprep_k<<<1, 128, 0, stream>>>(bn2g, bn2b, bn2m, bn2v, c2b, sc2, bi2, 128);
  bnprep_k<<<1, 256, 0, stream>>>(bn3g, bn3b, bn3m, bn3v, c3b, sc3, bi3, 256);

  conv1_k<<<(BB * 64 * 256 * 256) / 256, 256, 0, stream>>>(x, c1w, sc1, bi1, h1);
  conv_wmma(stream, h1, c2w, sc2, bi2, h2, 64, 256, 256, 128, 128, 128, 3, 2, 1, 1);
  conv_wmma(stream, h2, c3w, sc3, bi3, h3, 128, 128, 128, 256, 64, 64, 3, 2, 1, 1);
  posenc_k<<<(256 * 64 * 64) / 256, 256, 0, stream>>>(h3);
  conv_wmma(stream, h3, latw, nullptr, latb, lat, 256, 64, 64, 256, 64, 64, 1, 1, 0, 0);
  conv_wmma(stream, lat, smw, nullptr, smb, feat, 256, 64, 64, 256, 64, 64, 3, 1, 1, 0);

  // -------- decoder init --------
  qinit_k<<<(BB * NQ * DD + 255) / 256, 256, 0, stream>>>(qemb, qpos, queries);
  copy_k<<<(BB * NQ * 4 + 255) / 256, 256, 0, stream>>>(boxes, pbi, BB * NQ * 4);

  for (int l = 0; l < NL; ++l) {
    // reference point offsets: tanh(relu(Q@rp1+..)@rp2+..)
    gemm(stream, queries, rp1W, rp1b, t1, BB * NQ, DD, DD, 1, 1, 1.f);
    gemm(stream, t1, rp2W, rp2b, ro, BB * NQ, 2 * NP, DD, 1, 2, 1.f);
    // bilinear patch sampling
    sample_k<<<BB * NQ * NP * PS * PS, DD, 0, stream>>>(feat, ro, boxes, sbuf);
    // patch MLP (dominant GEMM: 6400 x 256 x 2304)
    gemm(stream, sbuf, pm1W + (size_t)l * DD * (PS * PS * DD), pm1b + l * DD, t2,
         BB * NQ * NP, DD, PS * PS * DD, 1, 1, 1.f);
    gemm(stream, t2, pm2W + (size_t)l * DD * DD, pm2b + l * DD, pv,
         BB * NQ * NP, DD, DD, 1, 0, 1.f);
    mean_k<<<BB * NQ, DD, 0, stream>>>(pv, kv);
    // attention
    gemm(stream, queries, qW + (size_t)l * DD * DD, qb + l * DD, qbuf,
         BB * NQ, DD, DD, 1, 0, 1.f);
    gemm(stream, qbuf, kv, nullptr, scores, NQ, NQ, DD, 1, 0, scale,
         BB, (long)NQ * DD, (long)NQ * DD, (long)NQ * NQ);
    softmax_k<<<(BB * NQ + 7) / 8, 256, 0, stream>>>(scores, BB * NQ, NQ);
    gemm(stream, scores, kv, nullptr, ctx, NQ, DD, NQ, 0, 0, 1.f,
         BB, (long)NQ * NQ, (long)NQ * DD, (long)NQ * DD);
    gemm(stream, ctx, outW + (size_t)l * DD * DD, outb + l * DD, outbuf,
         BB * NQ, DD, DD, 1, 0, 1.f);
    add_k<<<(BB * NQ * DD + 255) / 256, 256, 0, stream>>>(queries, outbuf, BB * NQ * DD);
    // box refinement
    gemm(stream, queries, bh1W, bh1b, t1, BB * NQ, DD, DD, 1, 1, 1.f);
    gemm(stream, t1, bh2W, bh2b, delta, BB * NQ, 4, DD, 1, 3, 1.f);
    refine_k<<<(BB * NQ * 4 + 255) / 256, 256, 0, stream>>>(boxes, delta, BB * NQ * 4);
  }

  // -------- heads --------
  gemm(stream, queries, clsW, clsb, out, BB * NQ, 1, DD, 1, 0, 1.f);          // logits [800]
  copy_k<<<(BB * NQ * 4 + 255) / 256, 256, 0, stream>>>(out + BB * NQ, boxes,
                                                        BB * NQ * 4);         // boxes [3200]
}
